// MultiHeadGAT_57543971832575
// MI455X (gfx1250) — matrix operand order repaired and checked by reading, not compiled, not run on hardware
//
#include <hip/hip_runtime.h>

#define GN     4096
#define GIN    256
#define GFH    64
#define GNH    4
#define GWORDS (GN / 32)    // 128 packed-adj words per row

// Padded LDS row stride for the B tile: 64B of data + 32B pad = 96B (48 halves).
// Keeps ds_load_b128 32B-aligned and limits bank conflicts to 2-way.
#define BROW   48

typedef __attribute__((ext_vector_type(16))) _Float16 v16h;
typedef __attribute__((ext_vector_type(8)))  _Float16 v8h;
typedef __attribute__((ext_vector_type(8)))  float    v8f;
typedef __attribute__((ext_vector_type(4)))  float    v4f;
typedef __attribute__((ext_vector_type(4)))  unsigned u32x4;
typedef __attribute__((ext_vector_type(8)))  int      i32x8;
typedef __attribute__((ext_vector_type(4)))  int      i32x4;

#if defined(__has_builtin)
#if __has_builtin(__builtin_amdgcn_tensor_load_to_lds)
#define HAVE_TDM 1
#else
#define HAVE_TDM 0
#endif
#else
#define HAVE_TDM 0
#endif

static __device__ __forceinline__ v8f wmma_f16(v16h a, v16h b, v8f c) {
  return __builtin_amdgcn_wmma_f32_16x16x32_f16(
      false, a, false, b, (short)0, c, false, false);
}

#if HAVE_TDM
// Tensor Data Mover: 2D tile load (tile_dim0 = 32 elems of 2B contiguous,
// tile_dim1 = 64 rows, tensor_dim0_stride = 4096 elems), with LDS padding of
// 8 dwords after every 16 dwords (row stride 64B -> 96B in LDS).
// This toolchain exposes the 6-arg builtin:
//   (u32x4 g0, i32x8 g1, i32x4 g2, i32x4 g3, i32x8 extra, i32 cpol)
static __device__ __forceinline__ void tdm_load_tile(unsigned lds_addr,
                                                     unsigned long long gaddr) {
  u32x4 g0;
  g0.x = 1u;                                               // count=1 (valid D#)
  g0.y = lds_addr;                                         // LDS byte address
  g0.z = (unsigned)gaddr;                                  // global_addr[31:0]
  g0.w = (unsigned)((gaddr >> 32) & 0x01FFFFFFull)         // global_addr[56:32]
       | (2u << 30);                                       // type = 2 ("image")
  i32x8 g1;
  g1[0] = (1 << 16)                                        // data_size = 2B
        | (1 << 20)                                        // pad_enable
        | (3 << 22)                                        // pad_interval: 16 dw
        | (7 << 25);                                       // pad_amount: 8 dw
  g1[1] = (int)((GN & 0xFFFF) << 16);                      // tensor_dim0 lo16
  g1[2] = (int)(((GN >> 16) & 0xFFFF) | (64 << 16));       // dim0 hi16 | dim1 lo16
  g1[3] = (int)(32u << 16);                                // tile_dim0 = 32
  g1[4] = 64;                                              // tile_dim1 = 64
  g1[5] = GN;                                              // dim0_stride lo32
  g1[6] = 0;
  g1[7] = 0;
  i32x4 z4 = {0, 0, 0, 0};                                 // groups 2/3: 2D only
  i32x8 z8 = {0, 0, 0, 0, 0, 0, 0, 0};
  __builtin_amdgcn_tensor_load_to_lds(g0, g1, z4, z4, z8, 0);
}
#endif

// Issue the staging of one 64x32 f16 B-tile (xpT rows f=0..63, cols j0..j0+31)
// into LDS buffer `buf` with 96B padded rows.
static __device__ __forceinline__ void stage_issue(_Float16 (*bbuf)[64 * BROW], int buf,
                                                   const _Float16* xpTh, int j0,
                                                   int tid, int wave) {
#if HAVE_TDM
  if (wave == 0) {
    tdm_load_tile((unsigned)(unsigned long long)&bbuf[buf][0],
                  (unsigned long long)(xpTh + j0));
  }
#else
  // Fallback: cooperative async global->LDS, one b128 lane-transfer per thread.
  const int f = tid >> 2, seg = tid & 3;
  unsigned la = (unsigned)(unsigned long long)&bbuf[buf][0]
              + (unsigned)(f * 96 + seg * 16);
  unsigned long long ga = (unsigned long long)(xpTh + (size_t)f * GN + j0)
                        + (unsigned)(seg * 16);
  asm volatile("global_load_async_to_lds_b128 %0, %1, off"
               :: "v"(la), "v"(ga) : "memory");
#endif
}

static __device__ __forceinline__ void stage_wait_pending1(int wave) {
#if HAVE_TDM
  if (wave == 0) __builtin_amdgcn_s_wait_tensorcnt(1);   // current tile done
#else
  asm volatile("s_wait_asynccnt 0x1" ::: "memory");
#endif
}

static __device__ __forceinline__ void stage_wait_all(int wave) {
#if HAVE_TDM
  if (wave == 0) __builtin_amdgcn_s_wait_tensorcnt(0);
#else
  asm volatile("s_wait_asynccnt 0x0" ::: "memory");
#endif
}

// ---------------------------------------------------------------------------
// Kernel 1: pack adjacency (+ self loops) into a bitmask via wave32 ballot.
// ---------------------------------------------------------------------------
__global__ __launch_bounds__(256) void k_pack_adj(const int* __restrict__ adj,
                                                  unsigned* __restrict__ bits) {
  size_t t = (size_t)blockIdx.x * 256 + threadIdx.x;
  int row = (int)(t >> 12);
  int col = (int)(t & (GN - 1));
  int pred = (adj[t] != 0) || (row == col);
  unsigned long long b = __ballot(pred);
  if ((threadIdx.x & 31u) == 0) bits[t >> 5] = (unsigned)b;
}

// ---------------------------------------------------------------------------
// Kernel 2: f32 -> f16 conversions: x -> xf16[n][d], W[h][d][f] -> wt[h][f][d].
// ---------------------------------------------------------------------------
__global__ __launch_bounds__(256) void k_cvt(const float* __restrict__ x,
                                             const float* __restrict__ w,
                                             _Float16* __restrict__ xf,
                                             _Float16* __restrict__ wt) {
  int t = blockIdx.x * 256 + threadIdx.x;
  if (t < GN * GIN) xf[t] = (_Float16)x[t];
  if (t < GNH * GFH * GIN) {
    int h   = t / (GFH * GIN);
    int rem = t - h * GFH * GIN;
    int f   = rem / GIN;
    int d   = rem - f * GIN;
    wt[t] = (_Float16)w[(h * GIN + d) * GFH + f];
  }
}

// ---------------------------------------------------------------------------
// Kernel 3: x_prime = x @ W_h + b via WMMA f16; emits xpT (f16, transposed)
// and xp32 (f32) for the score dots.
// ---------------------------------------------------------------------------
__global__ __launch_bounds__(256) void k_xprime(const _Float16* __restrict__ xf,
                                                const _Float16* __restrict__ wt,
                                                const float* __restrict__ bias,
                                                float* __restrict__ xp32,
                                                _Float16* __restrict__ xpT) {
  const int lane = threadIdx.x & 31;
  const int wave = threadIdx.x >> 5;
  const int head = blockIdx.y;
  const int tile = blockIdx.x * 8 + wave;
  const int row0 = tile * 16;
  const int hi   = lane >> 4;
  const int r    = lane & 15;

  const _Float16* xrow = xf + (size_t)(row0 + r) * GIN;
  const _Float16* wth  = wt + (size_t)head * GFH * GIN;

  v8f acc[4] = {v8f{}, v8f{}, v8f{}, v8f{}};

#pragma unroll
  for (int d0 = 0; d0 < GIN; d0 += 32) {
    v16h a = *(const v16h*)(xrow + d0 + 16 * hi);
#pragma unroll
    for (int fc = 0; fc < 4; ++fc) {
      v16h b = *(const v16h*)(wth + (size_t)(fc * 16 + r) * GIN + d0 + 16 * hi);
      acc[fc] = wmma_f16(a, b, acc[fc]);
    }
  }

#pragma unroll
  for (int fc = 0; fc < 4; ++fc) {
    const int f = fc * 16 + r;
    const float bv = bias[head * GFH + f];
#pragma unroll
    for (int k = 0; k < 8; ++k) {
      const int rowM = row0 + k + 8 * hi;
      const float v = acc[fc][k] + bv;
      xp32[((size_t)head * GN + rowM) * GFH + f] = v;
      xpT[((size_t)head * GFH + f) * GN + rowM]  = (_Float16)v;
    }
  }
}

// ---------------------------------------------------------------------------
// Kernel 4: s_src / s_dst score dots.
// ---------------------------------------------------------------------------
__global__ __launch_bounds__(256) void k_s(const float* __restrict__ xp32,
                                           const float* __restrict__ phis,
                                           float* __restrict__ s_src,
                                           float* __restrict__ s_dst) {
  int idx = blockIdx.x * 256 + threadIdx.x;
  int head = idx >> 12;
  const float* xr = xp32 + (size_t)idx * GFH;
  const float* ps = phis + head * 2 * GFH;
  const float* pd = ps + GFH;
  float a = 0.f, b = 0.f;
#pragma unroll
  for (int f = 0; f < GFH; ++f) {
    a += xr[f] * ps[f];
    b += xr[f] * pd[f];
  }
  s_src[idx] = a;
  s_dst[idx] = b;
}

// ---------------------------------------------------------------------------
// Kernel 5: fused masked softmax + P@V (flash-attention style).
//  - s_dst staged to LDS once per block (async global->LDS).
//  - B tiles staged per chunk via TDM (double buffered, overlapped with
//    compute; chunk 0 overlaps the row-max pass).
// ---------------------------------------------------------------------------
__global__ __launch_bounds__(256) void k_attn(const unsigned* __restrict__ adjbits,
                                              const float* __restrict__ s_src,
                                              const float* __restrict__ s_dst,
                                              const _Float16* __restrict__ xpT,
                                              float* __restrict__ out) {
  __shared__ float    sdsh[GN];              // 16 KB: s_dst for this head
  __shared__ _Float16 bbuf[2][64 * BROW];    // 2 x 6 KB: padded B tiles

  const int tid  = threadIdx.x;
  const int lane = tid & 31;
  const int wave = tid >> 5;
  const int head = blockIdx.y;
  const int tile = blockIdx.x * 8 + wave;
  const int row0 = tile * 16;
  const int hi   = lane >> 4;
  const int r    = lane & 15;
  const int row  = row0 + r;

  const float ssrc = s_src[head * GN + row];
  const unsigned* __restrict__ abrow = adjbits + (size_t)row * GWORDS;
  const _Float16* __restrict__ xpTh = xpT + (size_t)head * GFH * GN;

  // ---- async preload s_dst[head][:] into LDS (4 x b128 per thread) ----
  {
    unsigned lbase = (unsigned)(unsigned long long)sdsh;
    unsigned long long gbase = (unsigned long long)(s_dst + (size_t)head * GN);
#pragma unroll
    for (int it = 0; it < 4; ++it) {
      const unsigned off = (unsigned)((it * 256 + tid) * 16);
      unsigned la = lbase + off;
      unsigned long long ga = gbase + off;
      asm volatile("global_load_async_to_lds_b128 %0, %1, off"
                   :: "v"(la), "v"(ga) : "memory");
    }
    asm volatile("s_wait_asynccnt 0x0" ::: "memory");
  }
  __syncthreads();

  // Kick off the first B tile so it streams in during pass 1.
  stage_issue(bbuf, 0, xpTh, 0, tid, wave);

  // ---- Pass 1: masked row max (scores from LDS) ----
  float m = -3e38f;
  for (int c = 0; c < GWORDS; ++c) {
    __builtin_prefetch(abrow + c + 16, 0, 1);
    const unsigned wmask = abrow[c];
    const int j0 = c * 32;
    v4f s0 = *(const v4f*)(sdsh + j0 + 8 * hi);
    v4f s1 = *(const v4f*)(sdsh + j0 + 8 * hi + 4);
    v4f s2 = *(const v4f*)(sdsh + j0 + 16 + 8 * hi);
    v4f s3 = *(const v4f*)(sdsh + j0 + 16 + 8 * hi + 4);
    float sv[16];
#pragma unroll
    for (int e = 0; e < 4; ++e) {
      sv[e] = s0[e]; sv[4 + e] = s1[e]; sv[8 + e] = s2[e]; sv[12 + e] = s3[e];
    }
#pragma unroll
    for (int e = 0; e < 16; ++e) {
      const int K = (e < 8) ? (8 * hi + e) : (8 + 8 * hi + e);
      float S = ssrc + sv[e];
      S = (S < 0.f) ? 0.2f * S : S;
      m = fmaxf(m, ((wmask >> K) & 1u) ? S : -3e38f);
    }
  }
  m = fmaxf(m, __shfl_xor(m, 16, 32));

  // ---- Pass 2: exp + WMMA, B tiles double-buffered through LDS ----
  v8f acc[4] = {v8f{}, v8f{}, v8f{}, v8f{}};
  float zsum = 0.f;
  for (int c = 0; c < GWORDS; ++c) {
    if (c + 1 < GWORDS) {
      stage_issue(bbuf, (c + 1) & 1, xpTh, (c + 1) * 32, tid, wave);
      stage_wait_pending1(wave);
    } else {
      stage_wait_all(wave);
    }
    __syncthreads();                       // publish bbuf[c&1]

    const unsigned wmask = abrow[c];
    const int j0 = c * 32;
    v4f s0 = *(const v4f*)(sdsh + j0 + 8 * hi);
    v4f s1 = *(const v4f*)(sdsh + j0 + 8 * hi + 4);
    v4f s2 = *(const v4f*)(sdsh + j0 + 16 + 8 * hi);
    v4f s3 = *(const v4f*)(sdsh + j0 + 16 + 8 * hi + 4);
    float sv[16];
#pragma unroll
    for (int e = 0; e < 4; ++e) {
      sv[e] = s0[e]; sv[4 + e] = s1[e]; sv[8 + e] = s2[e]; sv[12 + e] = s3[e];
    }
    v16h a;
#pragma unroll
    for (int e = 0; e < 16; ++e) {
      const int K = (e < 8) ? (8 * hi + e) : (8 + 8 * hi + e);
      float S = ssrc + sv[e];
      S = (S < 0.f) ? 0.2f * S : S;
      const float p = ((wmask >> K) & 1u) ? __expf(S - m) : 0.f;
      zsum += p;
      a[e] = (_Float16)p;
    }
    const _Float16* bb = &bbuf[c & 1][0];
#pragma unroll
    for (int fc = 0; fc < 4; ++fc) {
      v8h blo = *(const v8h*)(bb + (fc * 16 + r) * BROW + 8 * hi);
      v8h bhi = *(const v8h*)(bb + (fc * 16 + r) * BROW + 16 + 8 * hi);
      v16h b = __builtin_shufflevector(blo, bhi, 0, 1, 2, 3, 4, 5, 6, 7,
                                       8, 9, 10, 11, 12, 13, 14, 15);
      acc[fc] = wmma_f16(a, b, acc[fc]);
    }
    __syncthreads();                       // protect bbuf[c&1] vs chunk c+2
  }

  // ---- Normalize and write out[n][head*64 + f] ----
  zsum += __shfl_xor(zsum, 16, 32);
#pragma unroll
  for (int k = 0; k < 8; ++k) {
    const int rr = k + 8 * hi;
    const float z = __shfl(zsum, rr, 32);
    const float inv = 1.0f / z;
    float* op = out + (size_t)(row0 + rr) * (GNH * GFH) + head * GFH;
    op[0 * 16 + r] = acc[0][k] * inv;
    op[1 * 16 + r] = acc[1][k] * inv;
    op[2 * 16 + r] = acc[2][k] * inv;
    op[3 * 16 + r] = acc[3][k] * inv;
  }
}

// ---------------------------------------------------------------------------
extern "C" void kernel_launch(void* const* d_in, const int* in_sizes, int n_in,
                              void* d_out, int out_size, void* d_ws, size_t ws_size,
                              hipStream_t stream) {
  const int*   adj  = (const int*)d_in[0];
  const float* x    = (const float*)d_in[1];
  const float* w    = (const float*)d_in[2];
  const float* bias = (const float*)d_in[3];
  const float* phis = (const float*)d_in[4];
  float* out = (float*)d_out;

  char* ws = (char*)d_ws;
  auto carve = [&](size_t bytes) -> char* {
    char* p = ws;
    ws += (bytes + 255) & ~(size_t)255;
    return p;
  };
  unsigned* adjbits = (unsigned*)carve((size_t)GN * GWORDS * 4);     // 2 MB
  _Float16* xf16    = (_Float16*)carve((size_t)GN * GIN * 2);        // 2 MB
  _Float16* wt      = (_Float16*)carve((size_t)GNH * GFH * GIN * 2); // 128 KB
  float*    xp32    = (float*)carve((size_t)GNH * GN * GFH * 4);     // 4 MB
  _Float16* xpT     = (_Float16*)carve((size_t)GNH * GFH * GN * 2);  // 2 MB
  float*    ssrc    = (float*)carve((size_t)GNH * GN * 4);           // 64 KB
  float*    sdst    = (float*)carve((size_t)GNH * GN * 4);           // 64 KB

  k_pack_adj<<<dim3((unsigned)(((size_t)GN * GN) / 256)), dim3(256), 0, stream>>>(adj, adjbits);
  k_cvt<<<dim3(GN * GIN / 256), dim3(256), 0, stream>>>(x, w, xf16, wt);
  k_xprime<<<dim3(32, GNH), dim3(256), 0, stream>>>(xf16, wt, bias, xp32, xpT);
  k_s<<<dim3(GNH * GN / 256), dim3(256), 0, stream>>>(xp32, phis, ssrc, sdst);
  k_attn<<<dim3(32, GNH), dim3(256), 0, stream>>>(adjbits, ssrc, sdst, xpT, out);
}
